// NKIExpertMLP_24970939859025
// MI455X (gfx1250) — matrix operand order repaired and checked by reading, not compiled, not run on hardware
//
#include <hip/hip_runtime.h>
#include <hip/hip_bf16.h>
#include <stdint.h>

#define TOKENS 8192
#define HIDDEN 2048
#define INTER  5632

// LDS row stride in bf16 elements: 32 data + 8 pad.
// - fragment reads: bank(m) = (m*20 + k/2) % 64 -> 16 distinct banks, conflict-free
// - 80-byte rows keep every 16-half boundary 16B-aligned for B128 LDS writes
#define LDP 40

typedef __attribute__((ext_vector_type(16))) __bf16 v16bf;
typedef __attribute__((ext_vector_type(8)))  float  v8f;

union FragBF {
    v16bf    v;
    uint32_t u[8];
};

__device__ __forceinline__ uint16_t bf16bits(float f) {
    union { __bf16 h; uint16_t u; } c;
    c.h = (__bf16)f;
    return c.u;
}
__device__ __forceinline__ uint32_t packbf(float lo, float hi) {
    union { __bf16 h[2]; uint32_t u; } c;
    c.h[0] = (__bf16)lo; c.h[1] = (__bf16)hi;   // v_cvt_pk_bf16_f32
    return c.u;
}

// CDNA5 async copy: 16 bytes global -> LDS per lane, tracked by ASYNCcnt.
__device__ __forceinline__ void async_b128(uint32_t lds_off, uint64_t gaddr) {
    asm volatile("global_load_async_to_lds_b128 %0, %1, off"
                 :: "v"(lds_off), "v"(gaddr) : "memory");
}
__device__ __forceinline__ void wait_async0() {
    asm volatile("s_wait_asynccnt 0x0" ::: "memory");
}

// ===========================================================================
// FAST PATH: one-time fp32->bf16 conversion (+ weight transpose), then
// all-bf16 GEMMs staged entirely with global_load_async_to_lds_b128.
// ===========================================================================

// ---- flat fp32 -> bf16 convert (8 elems/thread) ----
__global__ void __launch_bounds__(256)
cvt_bf16_kernel(const float* __restrict__ src, uint16_t* __restrict__ dst, int n8) {
    int i = blockIdx.x * blockDim.x + threadIdx.x;
    if (i >= n8) return;
    const float4* p = (const float4*)src + 2 * (size_t)i;
    float4 a = p[0], b = p[1];
    uint4 o;
    o.x = packbf(a.x, a.y); o.y = packbf(a.z, a.w);
    o.z = packbf(b.x, b.y); o.w = packbf(b.z, b.w);
    ((uint4*)dst)[i] = o;
}

// ---- tiled fp32 [R][C] -> bf16 transposed [C][R] ----
__global__ void __launch_bounds__(256)
cvtT_kernel(const float* __restrict__ src, uint16_t* __restrict__ dst, int R, int C) {
    __shared__ uint16_t tile[64][72];
    const int r0 = blockIdx.y * 64;
    const int c0 = blockIdx.x * 64;
    const int tr = threadIdx.x >> 2;          // 0..63
    const int tc = (threadIdx.x & 3) * 16;    // 16 floats / thread
    const float4* p = (const float4*)(src + (size_t)(r0 + tr) * C + c0 + tc);
#pragma unroll
    for (int q = 0; q < 4; ++q) {
        float4 f = p[q];
        tile[tr][tc + 4 * q + 0] = bf16bits(f.x);
        tile[tr][tc + 4 * q + 1] = bf16bits(f.y);
        tile[tr][tc + 4 * q + 2] = bf16bits(f.z);
        tile[tr][tc + 4 * q + 3] = bf16bits(f.w);
    }
    __syncthreads();
    const int wc = threadIdx.x >> 2;          // output row (= src col)
    const int wr = (threadIdx.x & 3) * 16;    // 16 along r
    uint16_t* o = dst + (size_t)(c0 + wc) * R + r0 + wr;
#pragma unroll
    for (int h = 0; h < 2; ++h) {
        union { uint16_t e[8]; uint4 v; } g;
#pragma unroll
        for (int j = 0; j < 8; ++j) g.e[j] = tile[wr + 8 * h + j][wc];
        *(uint4*)(o + 8 * h) = g.v;
    }
}

// ---- Kernel 1 (bf16): hidden = silu(x@Wg) * (x@Wu) ----
// Block 256(m) x 128(n), 512 threads = 16 waves, each wave 32x64.
// All staging via async global->LDS copies; weights pre-transposed [n][k].
__global__ void __launch_bounds__(512)
mlp1_bf16(const uint16_t* __restrict__ Xb,    // [TOKENS][HIDDEN]
          const uint16_t* __restrict__ WgT,   // [INTER][HIDDEN]  (gate, transposed)
          const uint16_t* __restrict__ WuT,   // [INTER][HIDDEN]  (up, transposed)
          uint16_t*       __restrict__ H) {   // [TOKENS][INTER]
    __shared__ uint16_t sA[2][256][LDP];
    __shared__ uint16_t sG[2][128][LDP];
    __shared__ uint16_t sU[2][128][LDP];

    const int tid  = threadIdx.x;
    const int lane = tid & 31;
    const int half = lane >> 4;
    const int l16  = lane & 15;
    const int wave = tid >> 5;     // 0..15
    const int wm   = wave >> 1;    // 0..7 : 32-row slab
    const int wn   = wave & 1;     // 0..1 : 64-col slab

    const int m0 = blockIdx.y * 256;
    const int n0 = blockIdx.x * 128;

    v8f accG[2][4];
    v8f accU[2][4];
#pragma unroll
    for (int i = 0; i < 2; ++i)
#pragma unroll
        for (int j = 0; j < 4; ++j) { accG[i][j] = 0.0f; accU[i][j] = 0.0f; }

    const int arow = tid >> 1;           // 0..255
    const int acol = (tid & 1) * 16;
    const int nrow = tid >> 2;           // 0..127
    const int ncol = (tid & 3) * 8;

    const uint16_t* ap = Xb  + (size_t)(m0 + arow) * HIDDEN + acol;
    const uint16_t* gp = WgT + (size_t)(n0 + nrow) * HIDDEN + ncol;
    const uint16_t* up = WuT + (size_t)(n0 + nrow) * HIDDEN + ncol;

    auto stage = [&](int b, int k0) {
#pragma unroll
        for (int q = 0; q < 2; ++q)
            async_b128((uint32_t)(uintptr_t)&sA[b][arow][acol + 8 * q],
                       (uint64_t)(uintptr_t)(ap + k0 + 8 * q));
        async_b128((uint32_t)(uintptr_t)&sG[b][nrow][ncol],
                   (uint64_t)(uintptr_t)(gp + k0));
        async_b128((uint32_t)(uintptr_t)&sU[b][nrow][ncol],
                   (uint64_t)(uintptr_t)(up + k0));
    };

    stage(0, 0);
    wait_async0();
    __syncthreads();

    const int NT = HIDDEN / 32;
    for (int t = 0; t < NT; ++t) {
        if (t + 1 < NT) stage((t + 1) & 1, (t + 1) * 32);

        const int b = t & 1;
        FragBF a[2];
#pragma unroll
        for (int i = 0; i < 2; ++i) {
            const int m = wm * 32 + i * 16 + l16;
#pragma unroll
            for (int v = 0; v < 8; ++v) {
                const int kb = v * 2 + half * 16;
                a[i].u[v] = *(const uint32_t*)&sA[b][m][kb];
            }
        }
#pragma unroll
        for (int j = 0; j < 4; ++j) {
            const int n = wn * 64 + j * 16 + l16;
            FragBF bg, bu;
#pragma unroll
            for (int v = 0; v < 8; ++v) {
                const int kb = v * 2 + half * 16;
                bg.u[v] = *(const uint32_t*)&sG[b][n][kb];
                bu.u[v] = *(const uint32_t*)&sU[b][n][kb];
            }
#pragma unroll
            for (int i = 0; i < 2; ++i) {
                accG[i][j] = __builtin_amdgcn_wmma_f32_16x16x32_bf16(
                    false, a[i].v, false, bg.v, (short)0, accG[i][j], false, false);
                accU[i][j] = __builtin_amdgcn_wmma_f32_16x16x32_bf16(
                    false, a[i].v, false, bu.v, (short)0, accU[i][j], false, false);
            }
        }

        wait_async0();
        __syncthreads();
    }

#pragma unroll
    for (int i = 0; i < 2; ++i) {
#pragma unroll
        for (int j = 0; j < 4; ++j) {
            const int col = n0 + wn * 64 + j * 16 + l16;
#pragma unroll
            for (int r = 0; r < 8; ++r) {
                const int row = m0 + wm * 32 + i * 16 + r + half * 8;
                float g = accG[i][j][r];
                float u = accU[i][j][r];
                float s = g / (1.0f + __expf(-g));
                H[(size_t)row * INTER + col] = bf16bits(s * u);
            }
        }
    }
}

// ---- Kernel 2 (bf16): out = hidden @ down ----
__global__ void __launch_bounds__(512)
mlp2_bf16(const uint16_t* __restrict__ H,     // [TOKENS][INTER]
          const uint16_t* __restrict__ WdT,   // [HIDDEN][INTER] (transposed)
          float*          __restrict__ Out) { // [TOKENS][HIDDEN]
    __shared__ uint16_t sA[2][256][LDP];
    __shared__ uint16_t sB[2][128][LDP];

    const int tid  = threadIdx.x;
    const int lane = tid & 31;
    const int half = lane >> 4;
    const int l16  = lane & 15;
    const int wave = tid >> 5;
    const int wm   = wave >> 1;
    const int wn   = wave & 1;

    const int m0 = blockIdx.y * 256;
    const int n0 = blockIdx.x * 128;

    v8f acc[2][4];
#pragma unroll
    for (int i = 0; i < 2; ++i)
#pragma unroll
        for (int j = 0; j < 4; ++j) acc[i][j] = 0.0f;

    const int arow = tid >> 1;
    const int acol = (tid & 1) * 16;
    const int nrow = tid >> 2;
    const int ncol = (tid & 3) * 8;

    const uint16_t* ap = H   + (size_t)(m0 + arow) * INTER + acol;
    const uint16_t* bp = WdT + (size_t)(n0 + nrow) * INTER + ncol;

    auto stage = [&](int b, int k0) {
#pragma unroll
        for (int q = 0; q < 2; ++q)
            async_b128((uint32_t)(uintptr_t)&sA[b][arow][acol + 8 * q],
                       (uint64_t)(uintptr_t)(ap + k0 + 8 * q));
        async_b128((uint32_t)(uintptr_t)&sB[b][nrow][ncol],
                   (uint64_t)(uintptr_t)(bp + k0));
    };

    stage(0, 0);
    wait_async0();
    __syncthreads();

    const int NT = INTER / 32;
    for (int t = 0; t < NT; ++t) {
        if (t + 1 < NT) stage((t + 1) & 1, (t + 1) * 32);

        const int b = t & 1;
        FragBF a[2];
#pragma unroll
        for (int i = 0; i < 2; ++i) {
            const int m = wm * 32 + i * 16 + l16;
#pragma unroll
            for (int v = 0; v < 8; ++v) {
                const int kb = v * 2 + half * 16;
                a[i].u[v] = *(const uint32_t*)&sA[b][m][kb];
            }
        }
#pragma unroll
        for (int j = 0; j < 4; ++j) {
            const int n = wn * 64 + j * 16 + l16;
            FragBF bB;
#pragma unroll
            for (int v = 0; v < 8; ++v) {
                const int kb = v * 2 + half * 16;
                bB.u[v] = *(const uint32_t*)&sB[b][n][kb];
            }
#pragma unroll
            for (int i = 0; i < 2; ++i) {
                acc[i][j] = __builtin_amdgcn_wmma_f32_16x16x32_bf16(
                    false, a[i].v, false, bB.v, (short)0, acc[i][j], false, false);
            }
        }

        wait_async0();
        __syncthreads();
    }

#pragma unroll
    for (int i = 0; i < 2; ++i) {
#pragma unroll
        for (int j = 0; j < 4; ++j) {
            const int col = n0 + wn * 64 + j * 16 + l16;
#pragma unroll
            for (int r = 0; r < 8; ++r) {
                const int row = m0 + wm * 32 + i * 16 + r + half * 8;
                Out[(size_t)row * HIDDEN + col] = acc[i][j][r];
            }
        }
    }
}

// ===========================================================================
// FALLBACK PATH (small workspace): round-2 convert-on-the-fly kernels.
// ===========================================================================
__global__ void __launch_bounds__(256)
gateup_silu_f32(const float* __restrict__ X, const float* __restrict__ W,
                uint16_t* __restrict__ H) {
    __shared__ uint16_t sA[2][128][LDP];
    __shared__ uint16_t sG[2][128][LDP];
    __shared__ uint16_t sU[2][128][LDP];

    const int tid  = threadIdx.x;
    const int lane = tid & 31;
    const int half = lane >> 4;
    const int l16  = lane & 15;
    const int wave = tid >> 5;
    const int wm   = wave >> 1;
    const int wn   = wave & 1;

    const int m0 = blockIdx.y * 128;
    const int n0 = blockIdx.x * 128;

    v8f accG[2][4];
    v8f accU[2][4];
#pragma unroll
    for (int i = 0; i < 2; ++i)
#pragma unroll
        for (int j = 0; j < 4; ++j) { accG[i][j] = 0.0f; accU[i][j] = 0.0f; }

    const int arow = tid >> 1;
    const int acol = (tid & 1) * 16;
    const int brow = tid >> 3;
    const int bcol = (tid & 7) * 16;

    const float* xp = X + (size_t)(m0 + arow) * HIDDEN + acol;
    const float* gp = W + (size_t)brow * (2 * INTER) + n0 + bcol;
    const float* up = gp + INTER;

    float4 fa[4], fg[4], fu[4];
    auto loadregs = [&](int k0) {
        const float4* pa = (const float4*)(xp + k0);
        const float4* pg = (const float4*)(gp + (size_t)k0 * (2 * INTER));
        const float4* pu = (const float4*)(up + (size_t)k0 * (2 * INTER));
#pragma unroll
        for (int q = 0; q < 4; ++q) { fa[q] = pa[q]; fg[q] = pg[q]; fu[q] = pu[q]; }
    };
    auto storelds = [&](int b) {
#pragma unroll
        for (int q = 0; q < 4; ++q) {
            *(uint32_t*)&sA[b][arow][acol + 4 * q]     = packbf(fa[q].x, fa[q].y);
            *(uint32_t*)&sA[b][arow][acol + 4 * q + 2] = packbf(fa[q].z, fa[q].w);
            const int nb = bcol + 4 * q;
            sG[b][nb + 0][brow] = bf16bits(fg[q].x);
            sG[b][nb + 1][brow] = bf16bits(fg[q].y);
            sG[b][nb + 2][brow] = bf16bits(fg[q].z);
            sG[b][nb + 3][brow] = bf16bits(fg[q].w);
            sU[b][nb + 0][brow] = bf16bits(fu[q].x);
            sU[b][nb + 1][brow] = bf16bits(fu[q].y);
            sU[b][nb + 2][brow] = bf16bits(fu[q].z);
            sU[b][nb + 3][brow] = bf16bits(fu[q].w);
        }
    };

    loadregs(0);
    storelds(0);
    __syncthreads();

    const int NT = HIDDEN / 32;
    for (int t = 0; t < NT; ++t) {
        if (t + 1 < NT) loadregs((t + 1) * 32);
        const int b = t & 1;
        FragBF a[2];
#pragma unroll
        for (int i = 0; i < 2; ++i) {
            const int m = wm * 32 + i * 16 + l16;
#pragma unroll
            for (int v = 0; v < 8; ++v) {
                const int kb = v * 2 + half * 16;
                a[i].u[v] = *(const uint32_t*)&sA[b][m][kb];
            }
        }
#pragma unroll
        for (int j = 0; j < 4; ++j) {
            const int n = wn * 64 + j * 16 + l16;
            FragBF bg, bu;
#pragma unroll
            for (int v = 0; v < 8; ++v) {
                const int kb = v * 2 + half * 16;
                bg.u[v] = *(const uint32_t*)&sG[b][n][kb];
                bu.u[v] = *(const uint32_t*)&sU[b][n][kb];
            }
#pragma unroll
            for (int i = 0; i < 2; ++i) {
                accG[i][j] = __builtin_amdgcn_wmma_f32_16x16x32_bf16(
                    false, a[i].v, false, bg.v, (short)0, accG[i][j], false, false);
                accU[i][j] = __builtin_amdgcn_wmma_f32_16x16x32_bf16(
                    false, a[i].v, false, bu.v, (short)0, accU[i][j], false, false);
            }
        }
        if (t + 1 < NT) storelds((t + 1) & 1);
        __syncthreads();
    }

#pragma unroll
    for (int i = 0; i < 2; ++i) {
#pragma unroll
        for (int j = 0; j < 4; ++j) {
            const int col = n0 + wn * 64 + j * 16 + l16;
#pragma unroll
            for (int r = 0; r < 8; ++r) {
                const int row = m0 + wm * 32 + i * 16 + r + half * 8;
                float g = accG[i][j][r];
                float u = accU[i][j][r];
                float s = g / (1.0f + __expf(-g));
                H[(size_t)row * INTER + col] = bf16bits(s * u);
            }
        }
    }
}

__global__ void __launch_bounds__(256)
down_f32(const uint16_t* __restrict__ H, const float* __restrict__ Wd,
         float* __restrict__ Out) {
    __shared__ uint16_t sA[2][128][LDP];
    __shared__ uint16_t sB[2][128][LDP];

    const int tid  = threadIdx.x;
    const int lane = tid & 31;
    const int half = lane >> 4;
    const int l16  = lane & 15;
    const int wave = tid >> 5;
    const int wm   = wave >> 1;
    const int wn   = wave & 1;

    const int m0 = blockIdx.y * 128;
    const int n0 = blockIdx.x * 128;

    v8f acc[2][4];
#pragma unroll
    for (int i = 0; i < 2; ++i)
#pragma unroll
        for (int j = 0; j < 4; ++j) acc[i][j] = 0.0f;

    const int arow = tid >> 1;
    const int acol = (tid & 1) * 16;
    const int brow = tid >> 3;
    const int bcol = (tid & 7) * 16;

    const uint16_t* hp = H  + (size_t)(m0 + arow) * INTER + acol;
    const float*    wp = Wd + (size_t)brow * HIDDEN + n0 + bcol;

    float4 fb[4];
    auto asyncA = [&](int b, int k0) {
#pragma unroll
        for (int q = 0; q < 2; ++q)
            async_b128((uint32_t)(uintptr_t)&sA[b][arow][acol + 8 * q],
                       (uint64_t)(uintptr_t)(hp + k0 + 8 * q));
    };
    auto loadB = [&](int k0) {
        const float4* p = (const float4*)(wp + (size_t)k0 * HIDDEN);
#pragma unroll
        for (int q = 0; q < 4; ++q) fb[q] = p[q];
    };
    auto storeB = [&](int b) {
#pragma unroll
        for (int q = 0; q < 4; ++q) {
            const int nb = bcol + 4 * q;
            sB[b][nb + 0][brow] = bf16bits(fb[q].x);
            sB[b][nb + 1][brow] = bf16bits(fb[q].y);
            sB[b][nb + 2][brow] = bf16bits(fb[q].z);
            sB[b][nb + 3][brow] = bf16bits(fb[q].w);
        }
    };

    asyncA(0, 0);
    loadB(0);
    storeB(0);
    wait_async0();
    __syncthreads();

    const int NT = INTER / 32;
    for (int t = 0; t < NT; ++t) {
        if (t + 1 < NT) { asyncA((t + 1) & 1, (t + 1) * 32); loadB((t + 1) * 32); }
        const int b = t & 1;
        FragBF a[2];
#pragma unroll
        for (int i = 0; i < 2; ++i) {
            const int m = wm * 32 + i * 16 + l16;
#pragma unroll
            for (int v = 0; v < 8; ++v) {
                const int kb = v * 2 + half * 16;
                a[i].u[v] = *(const uint32_t*)&sA[b][m][kb];
            }
        }
#pragma unroll
        for (int j = 0; j < 4; ++j) {
            const int n = wn * 64 + j * 16 + l16;
            FragBF bB;
#pragma unroll
            for (int v = 0; v < 8; ++v) {
                const int kb = v * 2 + half * 16;
                bB.u[v] = *(const uint32_t*)&sB[b][n][kb];
            }
#pragma unroll
            for (int i = 0; i < 2; ++i) {
                acc[i][j] = __builtin_amdgcn_wmma_f32_16x16x32_bf16(
                    false, a[i].v, false, bB.v, (short)0, acc[i][j], false, false);
            }
        }
        if (t + 1 < NT) storeB((t + 1) & 1);
        wait_async0();
        __syncthreads();
    }

#pragma unroll
    for (int i = 0; i < 2; ++i) {
#pragma unroll
        for (int j = 0; j < 4; ++j) {
            const int col = n0 + wn * 64 + j * 16 + l16;
#pragma unroll
            for (int r = 0; r < 8; ++r) {
                const int row = m0 + wm * 32 + i * 16 + r + half * 8;
                Out[(size_t)row * HIDDEN + col] = acc[i][j][r];
            }
        }
    }
}

// ===========================================================================
extern "C" void kernel_launch(void* const* d_in, const int* in_sizes, int n_in,
                              void* d_out, int out_size, void* d_ws, size_t ws_size,
                              hipStream_t stream) {
    (void)in_sizes; (void)n_in; (void)out_size;
    const float* x    = (const float*)d_in[0];
    const float* w_gu = (const float*)d_in[1];
    const float* w_d  = (const float*)d_in[2];
    float*       out  = (float*)d_out;

    // workspace layout (bytes)
    const size_t off_hid  = 0;                       // 92,274,688
    const size_t off_xbf  = 92274688;                // 33,554,432
    const size_t off_wguT = 125829120;               // 46,137,344
    const size_t off_wdT  = 171966464;               // 23,068,672
    const size_t need     = 195035136;

    uint16_t* hid = (uint16_t*)((char*)d_ws + off_hid);

    if (ws_size >= need) {
        uint16_t* xbf  = (uint16_t*)((char*)d_ws + off_xbf);
        uint16_t* wguT = (uint16_t*)((char*)d_ws + off_wguT);  // [2*INTER][HIDDEN]
        uint16_t* wdT  = (uint16_t*)((char*)d_ws + off_wdT);   // [HIDDEN][INTER]

        // one-time conversions / transposes
        const int n8x = TOKENS * HIDDEN / 8;
        cvt_bf16_kernel<<<dim3((n8x + 255) / 256), dim3(256), 0, stream>>>(x, xbf, n8x);
        cvtT_kernel<<<dim3(2 * INTER / 64, HIDDEN / 64), dim3(256), 0, stream>>>(
            w_gu, wguT, HIDDEN, 2 * INTER);
        cvtT_kernel<<<dim3(HIDDEN / 64, INTER / 64), dim3(256), 0, stream>>>(
            w_d, wdT, INTER, HIDDEN);

        mlp1_bf16<<<dim3(INTER / 128, TOKENS / 256), dim3(512), 0, stream>>>(
            xbf, wguT, wguT + (size_t)INTER * HIDDEN, hid);
        mlp2_bf16<<<dim3(HIDDEN / 128, TOKENS / 256), dim3(512), 0, stream>>>(
            hid, wdT, out);
    } else {
        dim3 block(256);
        gateup_silu_f32<<<dim3(INTER / 128, TOKENS / 128), block, 0, stream>>>(x, w_gu, hid);
        down_f32<<<dim3(HIDDEN / 128, TOKENS / 128), block, 0, stream>>>(hid, w_d, out);
    }
}
